// LyricsGeneratorAdvanced_10350871183855
// MI455X (gfx1250) — compile-verified, matrix operand order checked by reading, not run on hardware
//
#include <hip/hip_runtime.h>
#include <hip/hip_bf16.h>
#include <cstdint>

typedef __attribute__((ext_vector_type(16))) _Float16 v16h;
typedef __attribute__((ext_vector_type(8)))  _Float16 v8h;
typedef __attribute__((ext_vector_type(8)))  float    v8f;

namespace {
constexpr int kB = 8, kT = 256, kE = 512, kH = 512, kV = 32000;
constexpr int kBT = kB * kT;   // 2048
constexpr int kG  = 4 * kH;    // 2048 gate width
}

union V16H { v16h v; v8h h[2]; };

// ---- CDNA5-specific helpers -----------------------------------------------
// Async global->LDS copy (ASYNCcnt path). LDS byte address == low 32 bits of
// the flat shared pointer.
__device__ __forceinline__ void async_copy_b128(const _Float16* g, _Float16* l)
{
  unsigned lds = (unsigned)(uintptr_t)l;
  uint64_t ga  = (uint64_t)(uintptr_t)g;
  asm volatile("global_load_async_to_lds_b128 %0, %1, off"
               :: "v"(lds), "v"(ga) : "memory");
}
__device__ __forceinline__ void wait_async0()
{
#if __has_builtin(__builtin_amdgcn_s_wait_asynccnt)
  __builtin_amdgcn_s_wait_asynccnt(0);
#else
  asm volatile("s_wait_asynccnt 0x0" ::: "memory");
#endif
}
// Native gfx1250 tanh (TRANS32); add the §7.4 trans-op hazard delay.
__device__ __forceinline__ float fast_tanh(float x)
{
  float r;
  asm("v_tanh_f32 %0, %1\n\ts_delay_alu instid0(TRANS32_DEP_1)"
      : "=v"(r) : "v"(x));
  return r;
}
__device__ __forceinline__ float sigf(float x) { return 0.5f * fast_tanh(0.5f * x) + 0.5f; }

// Cooperative async stage of a 16 x K f16 strip (row stride lda) into padded LDS.
__device__ __forceinline__ void stage_strip(const _Float16* __restrict__ Ab,
                                            _Float16 (&alds)[16][520],
                                            int lda, int K)
{
  int cpr = K >> 3;                       // 16B chunks per row
  for (int ch = (int)threadIdx.x; ch < 2 * K; ch += (int)blockDim.x) {
    int row = ch / cpr, c = ch - row * cpr;
    async_copy_b128(Ab + (size_t)row * lda + c * 8, &alds[row][c * 8]);
  }
  wait_async0();
  __syncthreads();
}

// ---------------------------------------------------------------------------
// f16 WMMA GEMM:  D[M,N](f32) = A[M,K] * W[N,K]^T + bias
// Block = 8 waves = 8 consecutive 16x16 tiles sharing one mTile (all launches
// use nTiles % 8 == 0). A strip staged to LDS via async copy; W streamed.
// Fragment layout per CDNA5 ISA 7.12.2 (wave32).
// ---------------------------------------------------------------------------
__global__ void gemm_bt_f16(const _Float16* __restrict__ A,
                            const _Float16* __restrict__ W,
                            const float* __restrict__ bias,
                            float* __restrict__ D,
                            int M, int N, int K, int lda, int ldw, int ldd)
{
  __shared__ _Float16 alds[16][520];
  int nT = N >> 4;
  int tile0 = (int)blockIdx.x * 8;
  int mTile = tile0 / nT;
  stage_strip(A + (size_t)mTile * 16 * lda, alds, lda, K);
  if (mTile >= (M >> 4)) return;

  int lane  = (int)(threadIdx.x & 31);
  int nTile = (tile0 + (int)(threadIdx.x >> 5)) % nT;
  int r  = lane & 15;
  int kh = (lane >> 4) << 3;                     // 0 or 8
  const _Float16* pa = &alds[r][kh];
  const _Float16* pw = W + (size_t)(nTile * 16 + r) * ldw + kh;

  v8f acc = {};
  for (int kb = 0; kb < K; kb += 32) {
    __builtin_prefetch(pw + kb + 128, 0, 1);
    V16H a, b;
    a.h[0] = *(const v8h*)(pa + kb);
    a.h[1] = *(const v8h*)(pa + kb + 16);
    b.h[0] = *(const v8h*)(pw + kb);
    b.h[1] = *(const v8h*)(pw + kb + 16);
    acc = __builtin_amdgcn_wmma_f32_16x16x32_f16(false, a.v, false, b.v,
                                                 (short)0, acc, false, false);
  }
  int col  = nTile * 16 + r;
  float bv = bias ? bias[col] : 0.0f;
  int row0 = mTile * 16 + kh;
  float* pd = D + (size_t)row0 * ldd + col;
#pragma unroll
  for (int i = 0; i < 8; ++i) pd[(size_t)i * ldd] = acc[i] + bv;
}

// Batched D = A[M,K] * Bm[K,N]  (B row-major [K,N]; strided column loads)
__global__ void gemm_bn_f16_batched(const _Float16* __restrict__ A,
                                    const _Float16* __restrict__ Bm,
                                    float* __restrict__ D,
                                    int M, int N, int K, int lda, int ldb, int ldd,
                                    size_t sA, size_t sB, size_t sD)
{
  __shared__ _Float16 alds[16][520];
  A  += (size_t)blockIdx.y * sA;
  Bm += (size_t)blockIdx.y * sB;
  D  += (size_t)blockIdx.y * sD;

  int nT = N >> 4;
  int tile0 = (int)blockIdx.x * 8;
  int mTile = tile0 / nT;
  stage_strip(A + (size_t)mTile * 16 * lda, alds, lda, K);
  if (mTile >= (M >> 4)) return;

  int lane  = (int)(threadIdx.x & 31);
  int nTile = (tile0 + (int)(threadIdx.x >> 5)) % nT;
  int r  = lane & 15;
  int kh = (lane >> 4) << 3;
  const _Float16* pa = &alds[r][kh];

  v8f acc = {};
  for (int kb = 0; kb < K; kb += 32) {
    V16H a, b;
    a.h[0] = *(const v8h*)(pa + kb);
    a.h[1] = *(const v8h*)(pa + kb + 16);
    const _Float16* pb = Bm + (size_t)(kb + kh) * ldb + nTile * 16 + r;
#pragma unroll
    for (int j = 0; j < 8; ++j) b.h[0][j] = pb[(size_t)j * ldb];
    pb += (size_t)16 * ldb;
#pragma unroll
    for (int j = 0; j < 8; ++j) b.h[1][j] = pb[(size_t)j * ldb];
    acc = __builtin_amdgcn_wmma_f32_16x16x32_f16(false, a.v, false, b.v,
                                                 (short)0, acc, false, false);
  }
  int col  = nTile * 16 + r;
  int row0 = mTile * 16 + kh;
  float* pd = D + (size_t)row0 * ldd + col;
#pragma unroll
  for (int i = 0; i < 8; ++i) pd[(size_t)i * ldd] = acc[i];
}

// ---------------------------------------------------------------------------
// One LSTM timestep. 8 blocks x 256 thr. Block bi owns gate columns
// [bi*64, bi*64+64) of all 4 gates -> fused cell update in-block.
// h strip (16x512, rows>=8 zero) async-staged to LDS, shared by all 16 tiles.
// ---------------------------------------------------------------------------
__global__ void lstm_step(const float* __restrict__ Gpre,   // [B,4H] for this t
                          const _Float16* __restrict__ Wh,  // [4H,H]
                          const _Float16* __restrict__ hsrc,// [16,H] rows>=8 zero
                          _Float16* __restrict__ hdst,      // [16,H]
                          float* __restrict__ c,            // [B,H]
                          float* __restrict__ hout)         // [B,H] for this t
{
  __shared__ _Float16 alds[16][520];
  __shared__ float g_lds[4][8][64];
  stage_strip(hsrc, alds, kH, kH);

  const int wave = (int)(threadIdx.x >> 5), lane = (int)(threadIdx.x & 31);
  const int j0 = (int)blockIdx.x * 64;
  const int r = lane & 15, kh = (lane >> 4) << 3;
  const _Float16* pa = &alds[r][kh];

  for (int tt = wave; tt < 16; tt += 8) {            // 4 gates x 4 subtiles
    int gate = tt >> 2, sub = tt & 3;
    int colbase = gate * kH + j0 + sub * 16;
    const _Float16* pw = Wh + (size_t)(colbase + r) * kH + kh;
    v8f acc = {};
    for (int kb = 0; kb < kH; kb += 32) {
      V16H a, b;
      a.h[0] = *(const v8h*)(pa + kb);
      a.h[1] = *(const v8h*)(pa + kb + 16);
      b.h[0] = *(const v8h*)(pw + kb);
      b.h[1] = *(const v8h*)(pw + kb + 16);
      acc = __builtin_amdgcn_wmma_f32_16x16x32_f16(false, a.v, false, b.v,
                                                   (short)0, acc, false, false);
    }
    if (lane < 16) {
#pragma unroll
      for (int i = 0; i < 8; ++i) g_lds[gate][i][sub * 16 + r] = acc[i];
    }
  }
  __syncthreads();

  for (int idx = (int)threadIdx.x; idx < 512; idx += 256) {
    int b = idx >> 6, jj = idx & 63, j = j0 + jj;
    const float* gp = Gpre + (size_t)b * kG;
    float gi = g_lds[0][b][jj] + gp[j];
    float gf = g_lds[1][b][jj] + gp[kH + j];
    float gg = g_lds[2][b][jj] + gp[2 * kH + j];
    float go = g_lds[3][b][jj] + gp[3 * kH + j];
    float cn = sigf(gf) * c[(size_t)b * kH + j] + sigf(gi) * fast_tanh(gg);
    float hn = sigf(go) * fast_tanh(cn);
    c[(size_t)b * kH + j] = cn;
    hdst[(size_t)b * kH + j] = (_Float16)hn;
    hout[(size_t)b * kH + j] = hn;
  }
}

// LayerNorm over H; input [T,B,H], output rows (b*T+t) f32 + f16
__global__ void layernorm_k(const float* __restrict__ hin,
                            const float* __restrict__ gam,
                            const float* __restrict__ bet,
                            float* __restrict__ ln, _Float16* __restrict__ ln16)
{
  int row = (int)blockIdx.x;                 // b*T + t
  int b = row >> 8, t = row & 255;
  const float* src = hin + ((size_t)t * kB + b) * kH;
  __shared__ float red[256];
  int tid = (int)threadIdx.x;
  float x0 = src[tid], x1 = src[tid + 256];
  red[tid] = x0 + x1; __syncthreads();
  for (int s = 128; s > 0; s >>= 1) { if (tid < s) red[tid] += red[tid + s]; __syncthreads(); }
  float mu = red[0] * (1.0f / kH); __syncthreads();
  float d0 = x0 - mu, d1 = x1 - mu;
  red[tid] = d0 * d0 + d1 * d1; __syncthreads();
  for (int s = 128; s > 0; s >>= 1) { if (tid < s) red[tid] += red[tid + s]; __syncthreads(); }
  float rs = rsqrtf(red[0] * (1.0f / kH) + 1e-5f);
  float y0 = d0 * rs * gam[tid] + bet[tid];
  float y1 = d1 * rs * gam[tid + 256] + bet[tid + 256];
  float* o = ln + (size_t)row * kH;
  o[tid] = y0; o[tid + 256] = y1;
  _Float16* o16 = ln16 + (size_t)row * kH;
  o16[tid] = (_Float16)y0; o16[tid + 256] = (_Float16)y1;
}

// scores[b,t,k] = sum_h v[h]*tanh(keys[b,k,h]+qrys[b,t,h]); causal mask.
// Dominant transcendental kernel -> native v_tanh_f32.
__global__ void attn_scores_k(const float* __restrict__ keys,
                              const float* __restrict__ qrys,
                              const float* __restrict__ vvec,
                              float* __restrict__ scores)
{
  int t = (int)blockIdx.x, b = (int)blockIdx.y;
  __shared__ float q_lds[kH];
  const float* qr = qrys + ((size_t)b * kT + t) * kH;
  q_lds[threadIdx.x] = qr[threadIdx.x];
  q_lds[threadIdx.x + 256] = qr[threadIdx.x + 256];
  __syncthreads();
  int wave = (int)(threadIdx.x >> 5), lane = (int)(threadIdx.x & 31);
  float* srow = scores + ((size_t)b * kT + t) * kT;
  for (int k = wave; k < kT; k += 8) {
    if (k <= t) {
      const float* kr = keys + ((size_t)b * kT + k) * kH;
      float acc = 0.0f;
      for (int h = lane; h < kH; h += 32)
        acc += vvec[h] * fast_tanh(kr[h] + q_lds[h]);
      for (int m = 16; m >= 1; m >>= 1) acc += __shfl_xor(acc, m, 32);
      if (lane == 0) srow[k] = acc;
    } else if (lane == 0) {
      srow[k] = -1e10f;
    }
  }
}

__global__ void softmax_row_k(float* __restrict__ scores, _Float16* __restrict__ attn16)
{
  int row = (int)blockIdx.x;                 // b*T + t
  float* s = scores + (size_t)row * kT;
  __shared__ float red[256];
  int tid = (int)threadIdx.x;
  float x = s[tid];
  red[tid] = x; __syncthreads();
  for (int st = 128; st > 0; st >>= 1) { if (tid < st) red[tid] = fmaxf(red[tid], red[tid + st]); __syncthreads(); }
  float mx = red[0]; __syncthreads();
  float e = __expf(x - mx);
  red[tid] = e; __syncthreads();
  for (int st = 128; st > 0; st >>= 1) { if (tid < st) red[tid] += red[tid + st]; __syncthreads(); }
  float a = e / red[0];
  s[tid] = a;
  attn16[(size_t)row * kT + tid] = (_Float16)a;
}

// -------------------------- small utility kernels --------------------------
__global__ void cvt_f32_f16_k(const float* __restrict__ in, _Float16* __restrict__ out, size_t n)
{
  for (size_t i = blockIdx.x * blockDim.x + threadIdx.x; i < n; i += (size_t)gridDim.x * blockDim.x)
    out[i] = (_Float16)in[i];
}
__global__ void cvt_strided_f16_k(const float* __restrict__ in, _Float16* __restrict__ out,
                                  int rows, int cols, int ldin, int coloff)
{
  size_t n = (size_t)rows * cols;
  for (size_t i = blockIdx.x * blockDim.x + threadIdx.x; i < n; i += (size_t)gridDim.x * blockDim.x) {
    int r = (int)(i / cols), c = (int)(i - (size_t)r * cols);
    out[i] = (_Float16)in[(size_t)r * ldin + coloff + c];
  }
}
__global__ void cvt_x_tr_k(const float* __restrict__ x, _Float16* __restrict__ out)
{ // [B,T,E] -> f16 [(t*B+b), E]
  size_t n = (size_t)kBT * kE;
  for (size_t i = blockIdx.x * blockDim.x + threadIdx.x; i < n; i += (size_t)gridDim.x * blockDim.x) {
    int row = (int)(i >> 9), e = (int)(i & 511);
    int t = row >> 3, b = row & 7;
    out[i] = (_Float16)x[((size_t)b * kT + t) * kE + e];
  }
}
__global__ void add_vec_k(const float* a, const float* b, float* o, int n)
{
  int i = (int)(blockIdx.x * blockDim.x + threadIdx.x);
  if (i < n) o[i] = a[i] + b[i];
}
__global__ void zero_bytes_k(uint32_t* p, size_t nwords)
{
  for (size_t i = blockIdx.x * blockDim.x + threadIdx.x; i < nwords; i += (size_t)gridDim.x * blockDim.x)
    p[i] = 0u;
}

// ---------------------------------------------------------------------------
extern "C" void kernel_launch(void* const* d_in, const int* in_sizes, int n_in,
                              void* d_out, int out_size, void* d_ws, size_t ws_size,
                              hipStream_t stream)
{
  (void)in_sizes; (void)n_in; (void)out_size; (void)ws_size;
  const float* x      = (const float*)d_in[0];
  const float* Wih0   = (const float*)d_in[1];
  const float* Whh0   = (const float*)d_in[2];
  const float* b_ih0  = (const float*)d_in[3];
  const float* b_hh0  = (const float*)d_in[4];
  const float* Wih1   = (const float*)d_in[5];
  const float* Whh1   = (const float*)d_in[6];
  const float* b_ih1  = (const float*)d_in[7];
  const float* b_hh1  = (const float*)d_in[8];
  const float* ln_g   = (const float*)d_in[9];
  const float* ln_b   = (const float*)d_in[10];
  const float* attn_W = (const float*)d_in[11];
  const float* attn_b = (const float*)d_in[12];
  const float* vvec   = (const float*)d_in[13];
  const float* fc_W   = (const float*)d_in[14];
  const float* fc_b   = (const float*)d_in[15];
  float* out = (float*)d_out;

  char* ws = (char*)d_ws; size_t cur = 0;
  auto alloc = [&](size_t bytes) -> void* {
    void* p = ws + cur; cur = (cur + bytes + 255) & ~(size_t)255; return p;
  };
  _Float16* x16     = (_Float16*)alloc((size_t)kBT * kE * 2);
  _Float16* Wih0_16 = (_Float16*)alloc((size_t)kG * kE * 2);
  _Float16* Whh0_16 = (_Float16*)alloc((size_t)kG * kH * 2);
  _Float16* Wih1_16 = (_Float16*)alloc((size_t)kG * kH * 2);
  _Float16* Whh1_16 = (_Float16*)alloc((size_t)kG * kH * 2);
  _Float16* Wkey16  = (_Float16*)alloc((size_t)kH * kH * 2);
  _Float16* Wqry16  = (_Float16*)alloc((size_t)kH * kH * 2);
  _Float16* fcW16   = (_Float16*)alloc((size_t)kV * kH * 2);
  float*    bias0   = (float*)alloc((size_t)kG * 4);
  float*    bias1   = (float*)alloc((size_t)kG * 4);
  float*    Gpre    = (float*)alloc((size_t)kBT * kG * 4);
  _Float16* h16a    = (_Float16*)alloc((size_t)16 * kH * 2);
  _Float16* h16b    = (_Float16*)alloc((size_t)16 * kH * 2);
  float*    cbuf    = (float*)alloc((size_t)kB * kH * 4);
  float*    h1out   = (float*)alloc((size_t)kT * kB * kH * 4);
  _Float16* h1_16   = (_Float16*)alloc((size_t)kBT * kH * 2);
  float*    h2out   = (float*)alloc((size_t)kT * kB * kH * 4);
  float*    ln      = (float*)alloc((size_t)kBT * kH * 4);
  _Float16* ln16    = (_Float16*)alloc((size_t)kBT * kH * 2);
  float*    keys    = (float*)alloc((size_t)kBT * kH * 4);
  float*    qrys    = (float*)alloc((size_t)kBT * kH * 4);
  float*    scores  = (float*)alloc((size_t)kB * kT * kT * 4);
  _Float16* attn16  = (_Float16*)alloc((size_t)kB * kT * kT * 2);
  float*    ctx     = (float*)alloc((size_t)kBT * kH * 4);
  _Float16* ctx16   = (_Float16*)alloc((size_t)kBT * kH * 2);

  const int TH = 256;
  // --- weight / input conversions ---
  cvt_x_tr_k<<<2048, TH, 0, stream>>>(x, x16);
  cvt_f32_f16_k<<<2048, TH, 0, stream>>>(Wih0, Wih0_16, (size_t)kG * kE);
  cvt_f32_f16_k<<<2048, TH, 0, stream>>>(Whh0, Whh0_16, (size_t)kG * kH);
  cvt_f32_f16_k<<<2048, TH, 0, stream>>>(Wih1, Wih1_16, (size_t)kG * kH);
  cvt_f32_f16_k<<<2048, TH, 0, stream>>>(Whh1, Whh1_16, (size_t)kG * kH);
  cvt_strided_f16_k<<<1024, TH, 0, stream>>>(attn_W, Wkey16, kH, kH, 2 * kH, 0);
  cvt_strided_f16_k<<<1024, TH, 0, stream>>>(attn_W, Wqry16, kH, kH, 2 * kH, kH);
  cvt_f32_f16_k<<<4096, TH, 0, stream>>>(fc_W, fcW16, (size_t)kV * kH);
  add_vec_k<<<(kG + TH - 1) / TH, TH, 0, stream>>>(b_ih0, b_hh0, bias0, kG);
  add_vec_k<<<(kG + TH - 1) / TH, TH, 0, stream>>>(b_ih1, b_hh1, bias1, kG);

  // --- LSTM layer 0 ---
  zero_bytes_k<<<64, TH, 0, stream>>>((uint32_t*)h16a, (size_t)16 * kH / 2);
  zero_bytes_k<<<64, TH, 0, stream>>>((uint32_t*)h16b, (size_t)16 * kH / 2);
  zero_bytes_k<<<64, TH, 0, stream>>>((uint32_t*)cbuf, (size_t)kB * kH);
  gemm_bt_f16<<<2048, TH, 0, stream>>>(x16, Wih0_16, bias0, Gpre,
                                       kBT, kG, kE, kE, kE, kG);
  for (int t = 0; t < kT; ++t) {
    const _Float16* hs = (t & 1) ? h16b : h16a;
    _Float16*       hd = (t & 1) ? h16a : h16b;
    lstm_step<<<8, TH, 0, stream>>>(Gpre + (size_t)t * kB * kG, Whh0_16,
                                    hs, hd, cbuf, h1out + (size_t)t * kB * kH);
  }

  // --- LSTM layer 1 ---
  zero_bytes_k<<<64, TH, 0, stream>>>((uint32_t*)h16a, (size_t)16 * kH / 2);
  zero_bytes_k<<<64, TH, 0, stream>>>((uint32_t*)h16b, (size_t)16 * kH / 2);
  zero_bytes_k<<<64, TH, 0, stream>>>((uint32_t*)cbuf, (size_t)kB * kH);
  cvt_f32_f16_k<<<2048, TH, 0, stream>>>(h1out, h1_16, (size_t)kBT * kH);
  gemm_bt_f16<<<2048, TH, 0, stream>>>(h1_16, Wih1_16, bias1, Gpre,
                                       kBT, kG, kH, kH, kH, kG);
  for (int t = 0; t < kT; ++t) {
    const _Float16* hs = (t & 1) ? h16b : h16a;
    _Float16*       hd = (t & 1) ? h16a : h16b;
    lstm_step<<<8, TH, 0, stream>>>(Gpre + (size_t)t * kB * kG, Whh1_16,
                                    hs, hd, cbuf, h2out + (size_t)t * kB * kH);
  }

  // --- LayerNorm ---
  layernorm_k<<<kBT, TH, 0, stream>>>(h2out, ln_g, ln_b, ln, ln16);

  // --- attention projections (attn_b folded into keys) ---
  gemm_bt_f16<<<512, TH, 0, stream>>>(ln16, Wkey16, attn_b, keys,
                                      kBT, kH, kH, kH, kH, kH);
  gemm_bt_f16<<<512, TH, 0, stream>>>(ln16, Wqry16, nullptr, qrys,
                                      kBT, kH, kH, kH, kH, kH);

  // --- scores + softmax ---
  attn_scores_k<<<dim3(kT, kB), TH, 0, stream>>>(keys, qrys, vvec, scores);
  softmax_row_k<<<kBT, TH, 0, stream>>>(scores, attn16);

  // --- ctx[b] = attn[b] (TxT) x ln[b] (TxH), batched over b ---
  gemm_bn_f16_batched<<<dim3(64, kB), TH, 0, stream>>>(
      attn16, ln16, ctx, kT, kH, kT, kT, kH, kH,
      (size_t)kT * kT, (size_t)kT * kH, (size_t)kT * kH);
  cvt_f32_f16_k<<<2048, TH, 0, stream>>>(ctx, ctx16, (size_t)kBT * kH);

  // --- final FC: logits = ctx * fcW^T + fc_b  (store-bound) ---
  gemm_bt_f16<<<32000, TH, 0, stream>>>(ctx16, fcW16, fc_b, out,
                                        kBT, kV, kH, kH, kH, kV);
}
